// FlowGen_76149770158293
// MI455X (gfx1250) — matrix operand order repaired and checked by reading, not compiled, not run on hardware
//
#include <hip/hip_runtime.h>

typedef _Float16 v16h __attribute__((ext_vector_type(16)));
typedef _Float16 h8   __attribute__((ext_vector_type(8)));
typedef float    v8f  __attribute__((ext_vector_type(8)));

#define FG_WMMA(A, B, C) __builtin_amdgcn_wmma_f32_16x16x32_f16(false, A, false, B, (short)0, C, false, false)

// ---------------------------------------------------------------- helpers ---

__device__ __forceinline__ float fg_bil96(const float* __restrict__ img, int oy, int ox) {
    // bilinear, align_corners=True, 96 -> 48 grid: src = o * 95/47
    const float sc = 95.0f / 47.0f;
    float sy = oy * sc, sx = ox * sc;
    int y0 = (int)sy, x0 = (int)sx;               // oy,ox >= 0 -> trunc == floor
    int y1 = (y0 + 1 < 96) ? y0 + 1 : 95;
    int x1 = (x0 + 1 < 96) ? x0 + 1 : 95;
    float wy = sy - (float)y0, wx = sx - (float)x0;
    float a = img[y0 * 96 + x0], b = img[y0 * 96 + x1];
    float c = img[y1 * 96 + x0], d = img[y1 * 96 + x1];
    float top = a * (1.f - wx) + b * wx;
    float bot = c * (1.f - wx) + d * wx;
    return top * (1.f - wy) + bot * wy;
}

// A fragment: 16x32 f16.  lanes 0-15: M=lane, K={0..7,16..23}; lanes 16-31: M=lane-16, K={8..15,24..31}
__device__ __forceinline__ v16h fg_load_a(const _Float16* As, int mrow, int lane) {
    int l15 = lane & 15;
    int kb  = (lane & 16) ? 8 : 0;
    const _Float16* base = As + (mrow + l15) * 40;
    h8 lo = *(const h8*)(base + kb);
    h8 hi = *(const h8*)(base + 16 + kb);
    v16h a;
#pragma unroll
    for (int e = 0; e < 8; ++e) { a[e] = lo[e]; a[e + 8] = hi[e]; }
    return a;
}

// B fragment: 32x16 f16. lane&15 = column N; lanes 0-15 hold K={0..7,16..23}, lanes 16-31 hold K={8..15,24..31}
__device__ __forceinline__ v16h fg_load_b(const _Float16* Bs, int ncol, int lane) {
    int n  = lane & 15;
    int kb = (lane & 16) ? 8 : 0;
    const _Float16* base = Bs + (ncol + n) * 40 + kb;
    h8 lo = *(const h8*)(base);
    h8 hi = *(const h8*)(base + 16);
    v16h b;
#pragma unroll
    for (int e = 0; e < 8; ++e) { b[e] = lo[e]; b[e + 8] = hi[e]; }
    return b;
}

// ---------------------------------------------------------------- kernels ---

// mm[b,p] = (sum of 3x3 patch of bilinear(mask) == 0), ma[b,p] = bilinear(mask_all)
__global__ void fg_maskprep(const float* __restrict__ mask, const float* __restrict__ mask_all,
                            float* __restrict__ mm, float* __restrict__ ma) {
    int idx = blockIdx.x * blockDim.x + threadIdx.x;
    if (idx >= 2 * 2304) return;
    int b = idx / 2304, p = idx % 2304;
    int i = p / 48, j = p % 48;
    const float* mk  = mask + b * 96 * 96;
    const float* mka = mask_all + b * 96 * 96;
    ma[idx] = fg_bil96(mka, i, j);
    float s = 0.f;
    for (int ky = 0; ky < 3; ++ky)
        for (int kx = 0; kx < 3; ++kx) {
            int ii = i - 1 + ky, jj = j - 1 + kx;
            if (ii >= 0 && ii < 48 && jj >= 0 && jj < 48) s += fg_bil96(mk, ii, jj);
        }
    mm[idx] = (s == 0.0f) ? 1.0f : 0.0f;
}

// W[p][ch*9+t] = raw 3x3 patch of x2 (f16); Wn = same patch * 1/max(||patch||,1e-4)
__global__ void fg_build_w(const float* __restrict__ x2b, _Float16* __restrict__ W,
                           _Float16* __restrict__ Wn) {
    __shared__ float red[128];
    int p = blockIdx.x;
    int ch = threadIdx.x;                         // 128 threads
    int i = p / 48, j = p % 48;
    const float* xc = x2b + ch * 48 * 48;
    float vals[9];
    float ss = 0.f;
#pragma unroll
    for (int ky = 0; ky < 3; ++ky)
#pragma unroll
        for (int kx = 0; kx < 3; ++kx) {
            int y = i - 1 + ky, x = j - 1 + kx;
            float v = (y >= 0 && y < 48 && x >= 0 && x < 48) ? xc[y * 48 + x] : 0.f;
            vals[ky * 3 + kx] = v;
            ss += v * v;
            W[(size_t)p * 1152 + ch * 9 + ky * 3 + kx] = (_Float16)v;
        }
    red[ch] = ss;
    __syncthreads();
    for (int s = 64; s > 0; s >>= 1) {
        if (ch < s) red[ch] += red[ch + s];
        __syncthreads();
    }
    float inv = 1.0f / fmaxf(sqrtf(red[0]), 1e-4f);
#pragma unroll
    for (int t = 0; t < 9; ++t)
        Wn[(size_t)p * 1152 + ch * 9 + t] = (_Float16)(vals[t] * inv);
}

// S[p,c] = W[p] . Wn[c];  M=N=2304, K=1152
// block tile 256(M)x64(N)x32(K), 8 waves, wave tile 32x64 (8 WMMA / K-step)
__global__ __launch_bounds__(256) void fg_gemm_scores(const _Float16* __restrict__ W,
                                                      const _Float16* __restrict__ Wn,
                                                      float* __restrict__ S) {
    __shared__ alignas(16) _Float16 As[256 * 40];
    __shared__ alignas(16) _Float16 Bs[64 * 40];
    int tid = threadIdx.x;
    int lane = tid & 31, wv = tid >> 5;
    int Mb = blockIdx.x * 256, Nb = blockIdx.y * 64;
    v8f acc00 = {}, acc01 = {}, acc02 = {}, acc03 = {};
    v8f acc10 = {}, acc11 = {}, acc12 = {}, acc13 = {};
    for (int k0 = 0; k0 < 1152; k0 += 32) {
        __syncthreads();
#pragma unroll
        for (int it = 0; it < 4; ++it) {          // A: 256x32 halves in 16B chunks
            int cidx = it * 256 + tid;
            int m = cidx >> 2, ko = (cidx & 3) * 8;
            *(h8*)(As + m * 40 + ko) = *(const h8*)(W + (size_t)(Mb + m) * 1152 + k0 + ko);
        }
        {                                         // B: 64x32 halves (pre-normalized rows)
            int n = tid >> 2, ko = (tid & 3) * 8;
            *(h8*)(Bs + n * 40 + ko) = *(const h8*)(Wn + (size_t)(Nb + n) * 1152 + k0 + ko);
        }
        __syncthreads();
        v16h a0 = fg_load_a(As, wv * 32, lane);
        v16h a1 = fg_load_a(As, wv * 32 + 16, lane);
        v16h b0 = fg_load_b(Bs, 0, lane);
        acc00 = FG_WMMA(a0, b0, acc00);
        acc10 = FG_WMMA(a1, b0, acc10);
        v16h b1 = fg_load_b(Bs, 16, lane);
        acc01 = FG_WMMA(a0, b1, acc01);
        acc11 = FG_WMMA(a1, b1, acc11);
        v16h b2 = fg_load_b(Bs, 32, lane);
        acc02 = FG_WMMA(a0, b2, acc02);
        acc12 = FG_WMMA(a1, b2, acc12);
        v16h b3 = fg_load_b(Bs, 48, lane);
        acc03 = FG_WMMA(a0, b3, acc03);
        acc13 = FG_WMMA(a1, b3, acc13);
    }
    int rowB = Mb + wv * 32 + ((lane & 16) ? 8 : 0);
    int colB = Nb + (lane & 15);
#pragma unroll
    for (int r = 0; r < 8; ++r) {
        size_t ro = (size_t)(rowB + r) * 2304 + colB;
        S[ro +  0] = acc00[r];
        S[ro + 16] = acc01[r];
        S[ro + 32] = acc02[r];
        S[ro + 48] = acc03[r];
        size_t r1 = (size_t)(rowB + 16 + r) * 2304 + colB;
        S[r1 +  0] = acc10[r];
        S[r1 + 16] = acc11[r];
        S[r1 + 32] = acc12[r];
        S[r1 + 48] = acc13[r];
    }
}

// masked softmax over c (SCALE=10) + 4-neighbor boost + remask + clip -> f16
__global__ __launch_bounds__(256) void fg_softmax(const float* __restrict__ S,
                                                  const float* __restrict__ mm,
                                                  const float* __restrict__ ma,
                                                  _Float16* __restrict__ Yh) {
    __shared__ float red[256];
    int p = blockIdx.x;
    int t = threadIdx.x;
    float map = ma[p];
    int pi = p / 48, pj = p % 48;
    float mx = -3.4e38f;
    for (int c = t; c < 2304; c += 256) {
        float v = S[(size_t)p * 2304 + c] * mm[c] * map;
        mx = fmaxf(mx, v);
    }
    red[t] = mx; __syncthreads();
    for (int s = 128; s > 0; s >>= 1) { if (t < s) red[t] = fmaxf(red[t], red[t + s]); __syncthreads(); }
    mx = red[0]; __syncthreads();
    float sum = 0.f;
    for (int c = t; c < 2304; c += 256) {
        float v = S[(size_t)p * 2304 + c] * mm[c] * map;
        sum += __expf((v - mx) * 10.0f);
    }
    red[t] = sum; __syncthreads();
    for (int s = 128; s > 0; s >>= 1) { if (t < s) red[t] += red[t + s]; __syncthreads(); }
    float inv = 1.0f / red[0];
    for (int c = t; c < 2304; c += 256) {
        float v = S[(size_t)p * 2304 + c] * mm[c] * map;
        float z = __expf((v - mx) * 10.0f) * inv;
        int ci = c / 48, cj = c % 48;
        int d = (ci > pi ? ci - pi : pi - ci) + (cj > pj ? cj - pj : pj - cj);
        if (d == 1) z *= 1.5f;
        z *= mm[c] * map;
        z = fmaxf(z, 1e-8f);
        Yh[(size_t)p * 2304 + c] = (_Float16)z;
    }
}

// RWt[m][c] (N-major, m = ch*16+ky*4+kx): 4x4/stride-2 patches of x1 (f16), transposed
__global__ void fg_build_rw(const float* __restrict__ x1b, _Float16* __restrict__ RWt) {
    int idx = blockIdx.x * blockDim.x + threadIdx.x;     // 2048*2304 exactly
    int m = idx / 2304;
    int c = idx % 2304;
    int ch = m >> 4, kk = m & 15;
    int ky = kk >> 2, kx = kk & 3;
    int i = c / 48, j = c % 48;
    int Y = 2 * i - 1 + ky, X = 2 * j - 1 + kx;
    float v = (Y >= 0 && Y < 96 && X >= 0 && X < 96) ? x1b[(ch * 96 + Y) * 96 + X] : 0.f;
    RWt[idx] = (_Float16)v;
}

// OP[p, n] = sum_c Yh[p,c] * RWt[n,c];  M=2304, N=2048, K=2304
__global__ __launch_bounds__(256) void fg_gemm_out(const _Float16* __restrict__ Yh,
                                                   const _Float16* __restrict__ RWt,
                                                   float* __restrict__ OP) {
    __shared__ alignas(16) _Float16 As[256 * 40];
    __shared__ alignas(16) _Float16 Bs[64 * 40];
    int tid = threadIdx.x;
    int lane = tid & 31, wv = tid >> 5;
    int Mb = blockIdx.x * 256, Nb = blockIdx.y * 64;
    v8f acc00 = {}, acc01 = {}, acc02 = {}, acc03 = {};
    v8f acc10 = {}, acc11 = {}, acc12 = {}, acc13 = {};
    for (int k0 = 0; k0 < 2304; k0 += 32) {
        __syncthreads();
#pragma unroll
        for (int it = 0; it < 4; ++it) {          // A: 256x32 halves in 16B chunks
            int cidx = it * 256 + tid;
            int m = cidx >> 2, ko = (cidx & 3) * 8;
            *(h8*)(As + m * 40 + ko) = *(const h8*)(Yh + (size_t)(Mb + m) * 2304 + k0 + ko);
        }
        {                                         // B: 64x32 halves (RWt rows are contiguous in K)
            int n = tid >> 2, ko = (tid & 3) * 8;
            *(h8*)(Bs + n * 40 + ko) = *(const h8*)(RWt + (size_t)(Nb + n) * 2304 + k0 + ko);
        }
        __syncthreads();
        v16h a0 = fg_load_a(As, wv * 32, lane);
        v16h a1 = fg_load_a(As, wv * 32 + 16, lane);
        v16h b0 = fg_load_b(Bs, 0, lane);
        acc00 = FG_WMMA(a0, b0, acc00);
        acc10 = FG_WMMA(a1, b0, acc10);
        v16h b1 = fg_load_b(Bs, 16, lane);
        acc01 = FG_WMMA(a0, b1, acc01);
        acc11 = FG_WMMA(a1, b1, acc11);
        v16h b2 = fg_load_b(Bs, 32, lane);
        acc02 = FG_WMMA(a0, b2, acc02);
        acc12 = FG_WMMA(a1, b2, acc12);
        v16h b3 = fg_load_b(Bs, 48, lane);
        acc03 = FG_WMMA(a0, b3, acc03);
        acc13 = FG_WMMA(a1, b3, acc13);
    }
    int rowB = Mb + wv * 32 + ((lane & 16) ? 8 : 0);
    int colB = Nb + (lane & 15);
#pragma unroll
    for (int r = 0; r < 8; ++r) {
        size_t ro = (size_t)(rowB + r) * 2048 + colB;
        OP[ro +  0] = acc00[r];
        OP[ro + 16] = acc01[r];
        OP[ro + 32] = acc02[r];
        OP[ro + 48] = acc03[r];
        size_t r1 = (size_t)(rowB + 16 + r) * 2048 + colB;
        OP[r1 +  0] = acc10[r];
        OP[r1 + 16] = acc11[r];
        OP[r1 + 32] = acc12[r];
        OP[r1 + 48] = acc13[r];
    }
}

// overlap-add: y[ch,Y,X] = 0.25 * sum over coarse (i,j) with ky=Y+1-2i, kx=X+1-2j in [0,4)
__global__ void fg_gather(const float* __restrict__ OP, float* __restrict__ yb) {
    int idx = blockIdx.x * blockDim.x + threadIdx.x;     // 128*96*96 exactly
    int ch = idx / (96 * 96);
    int r = idx % (96 * 96);
    int Y = r / 96, X = r % 96;
    int ilo = (Y - 1) >> 1; if (ilo < 0) ilo = 0;
    int ihi = (Y + 1) >> 1; if (ihi > 47) ihi = 47;
    int jlo = (X - 1) >> 1; if (jlo < 0) jlo = 0;
    int jhi = (X + 1) >> 1; if (jhi > 47) jhi = 47;
    float s = 0.f;
    for (int i = ilo; i <= ihi; ++i) {
        int ky = Y + 1 - 2 * i;
        for (int j = jlo; j <= jhi; ++j) {
            int kx = X + 1 - 2 * j;
            s += OP[(size_t)(i * 48 + j) * 2048 + ch * 16 + ky * 4 + kx];
        }
    }
    yb[idx] = 0.25f * s;
}

// 4 dilated 3x3 convs (rates 1,2,4,8), 16 out-ch each, + bias, ReLU, concat on channel
__global__ void fg_final_conv(const float* __restrict__ y, const float* __restrict__ cw,
                              const float* __restrict__ cb, float* __restrict__ out) {
    int idx = blockIdx.x * blockDim.x + threadIdx.x;     // 2*64*9216 exactly
    int b  = idx / (64 * 9216);
    int r  = idx % (64 * 9216);
    int oc = r / 9216;
    int rr = r % 9216;
    int Y = rr / 96, X = rr % 96;
    int g = oc >> 4;
    int rate = 1 << g;                                   // RATES = 1,2,4,8
    float acc = cb[oc];
    const float* yb = y + (size_t)b * 128 * 9216;
    const float* wb = cw + (size_t)oc * 128 * 9;
    for (int ch = 0; ch < 128; ++ch) {
        const float* yc = yb + ch * 9216;
        const float* wp = wb + ch * 9;
#pragma unroll
        for (int ky = 0; ky < 3; ++ky) {
            int yyp = Y + rate * (ky - 1);
            if (yyp < 0 || yyp >= 96) continue;
#pragma unroll
            for (int kx = 0; kx < 3; ++kx) {
                int xxp = X + rate * (kx - 1);
                if (xxp < 0 || xxp >= 96) continue;
                acc += yc[yyp * 96 + xxp] * wp[ky * 3 + kx];
            }
        }
    }
    out[idx] = fmaxf(acc, 0.f);
}

// ------------------------------------------------------------------ launch ---

extern "C" void kernel_launch(void* const* d_in, const int* in_sizes, int n_in,
                              void* d_out, int out_size, void* d_ws, size_t ws_size,
                              hipStream_t stream) {
    (void)in_sizes; (void)n_in; (void)out_size; (void)ws_size;
    const float* x1       = (const float*)d_in[0];
    const float* x2       = (const float*)d_in[1];
    const float* mask     = (const float*)d_in[2];
    const float* mask_all = (const float*)d_in[3];
    const float* conv_w   = (const float*)d_in[4];
    const float* conv_b   = (const float*)d_in[5];
    float* out = (float*)d_out;

    char* ws = (char*)d_ws;
    size_t off = 0;
    auto alloc = [&](size_t bytes) {
        char* p = ws + off;
        off = (off + bytes + 255) & ~(size_t)255;
        return p;
    };
    float*    mm   = (float*)alloc((size_t)2 * 2304 * 4);
    float*    ma   = (float*)alloc((size_t)2 * 2304 * 4);
    float*    yy   = (float*)alloc((size_t)2 * 128 * 96 * 96 * 4);
    _Float16* W    = (_Float16*)alloc((size_t)2304 * 1152 * 2);   // reused per batch
    _Float16* Wn   = (_Float16*)alloc((size_t)2304 * 1152 * 2);
    float*    S    = (float*)alloc((size_t)2304 * 2304 * 4);
    _Float16* Yh   = (_Float16*)alloc((size_t)2304 * 2304 * 2);
    _Float16* RWt  = (_Float16*)alloc((size_t)2048 * 2304 * 2);
    float*    OP   = (float*)alloc((size_t)2304 * 2048 * 4);

    fg_maskprep<<<(2 * 2304 + 255) / 256, 256, 0, stream>>>(mask, mask_all, mm, ma);

    for (int b = 0; b < 2; ++b) {
        const float* x1b = x1 + (size_t)b * 128 * 96 * 96;
        const float* x2b = x2 + (size_t)b * 128 * 48 * 48;
        fg_build_w<<<2304, 128, 0, stream>>>(x2b, W, Wn);
        fg_gemm_scores<<<dim3(9, 36), 256, 0, stream>>>(W, Wn, S);
        fg_softmax<<<2304, 256, 0, stream>>>(S, mm + b * 2304, ma + b * 2304, Yh);
        fg_build_rw<<<18432, 256, 0, stream>>>(x1b, RWt);
        fg_gemm_out<<<dim3(9, 32), 256, 0, stream>>>(Yh, RWt, OP);
        fg_gather<<<4608, 256, 0, stream>>>(OP, yy + (size_t)b * 128 * 9216);
    }
    fg_final_conv<<<4608, 256, 0, stream>>>(yy, conv_w, conv_b, out);
}
